// EgoProximityAgentAttention_78288663872282
// MI455X (gfx1250) — matrix operand order repaired and checked by reading, not compiled, not run on hardware
//
#include <hip/hip_runtime.h>
#include <hip/hip_bf16.h>

// ---------------------------------------------------------------------------
// EgoProximityAgentAttention — CDNA5 (gfx1250, wave32)
// B=8, N=512, D=1024, H=16, HD=64, K in [8,12]
// f16 WMMA GEMMs (fp32 accumulate) with async global->LDS staging of B tiles.
// ---------------------------------------------------------------------------

typedef __attribute__((ext_vector_type(16))) _Float16 v16h;
typedef __attribute__((ext_vector_type(8)))  _Float16 v8h;
typedef __attribute__((ext_vector_type(8)))  float    v8f;

constexpr int cB = 8, cN = 512, cD = 1024, cH = 16, cHD = 64;
constexpr int cKMAX = 12, cKDEF = 8, cDM = 256;            // Dm = D/4
constexpr int cM = cB * cN;                                 // 4096 token rows
constexpr float cPROX = 20.0f;

// ---------------- workspace layout (bytes, all 256-aligned) ----------------
constexpr size_t OFF_K    = 0;                                   // int
constexpr size_t OFF_TK   = 256;                                 // int[B][13]
constexpr size_t OFF_NBR  = 1024;                                // int[M*KMAX]
constexpr size_t OFF_BIAS = OFF_NBR  + (size_t)cM * cKMAX * 4;   // f32[M*KMAX*H]
constexpr size_t OFF_QP   = OFF_BIAS + (size_t)cM * cKMAX * cH * 4;
constexpr size_t OFF_KP   = OFF_QP   + (size_t)cM * cD * 4;      // f32 K proj
constexpr size_t OFF_VP   = OFF_KP   + (size_t)cM * cD * 4;      // f32 V proj
constexpr size_t OFF_AH   = OFF_VP   + (size_t)cM * cD * 4;      // f16 tokens
constexpr size_t OFF_WQH  = OFF_AH   + (size_t)cM * cD * 2;      // f16 Wq
constexpr size_t OFF_WKH  = OFF_WQH  + (size_t)cD * cD * 2;      // f16 Wk
constexpr size_t OFF_WVH  = OFF_WKH  + (size_t)cD * cD * 2;      // f16 Wv
constexpr size_t OFF_WOH  = OFF_WVH  + (size_t)cD * cD * 2;      // f16 Wo
constexpr size_t OFF_AOH  = OFF_WOH  + (size_t)cD * cD * 2;      // f16 attn out

// ---------------------------------------------------------------------------
// 0) fp32 -> f16 pre-conversion (one shot; removes cvt from GEMM hot loop)
// ---------------------------------------------------------------------------
__global__ __launch_bounds__(256) void f32_to_f16_kernel(
    const float* __restrict__ src, _Float16* __restrict__ dst, int n) {
  const int i = (blockIdx.x * 256 + threadIdx.x) * 8;
  if (i >= n) return;
  const float4 x0 = *(const float4*)(src + i);
  const float4 x1 = *(const float4*)(src + i + 4);
  v8h o;
  o[0] = (_Float16)x0.x; o[1] = (_Float16)x0.y;
  o[2] = (_Float16)x0.z; o[3] = (_Float16)x0.w;
  o[4] = (_Float16)x1.x; o[5] = (_Float16)x1.y;
  o[6] = (_Float16)x1.z; o[7] = (_Float16)x1.w;
  *(v8h*)(dst + i) = o;
}

// ---------------------------------------------------------------------------
// 1) meta: K selection + per-batch sorted top-13 smallest distances
// ---------------------------------------------------------------------------
__global__ __launch_bounds__(256) void meta_kernel(
    const float* __restrict__ dist, const float* __restrict__ speed,
    int* __restrict__ Kout, int* __restrict__ tkidx) {
  __shared__ float red[256];
  const int t = threadIdx.x;

  float cnt = 0.f;
  for (int i = t; i < cB * cN; i += 256) cnt += (dist[i] < cPROX) ? 1.f : 0.f;
  red[t] = cnt; __syncthreads();
  for (int s = 128; s > 0; s >>= 1) { if (t < s) red[t] += red[t + s]; __syncthreads(); }
  const float total_close = red[0];
  __syncthreads();

  red[t] = (t < cB) ? speed[t] : 0.f; __syncthreads();
  for (int s = 128; s > 0; s >>= 1) { if (t < s) red[t] += red[t + s]; __syncthreads(); }

  if (t == 0) {
    const float avg_speed   = red[0] / (float)cB;
    const float avg_density = total_close / (float)(cB * cN);
    int K = cKDEF;
    if (avg_speed > 15.0f)  K = min(K + 1, cKMAX);
    if (avg_density > 0.5f) K = min(K + 1, cKMAX);
    *Kout = min(K, cN - 1);
  }

  if (t < cB) {   // sorted ascending, ties -> lower index (matches lax.top_k)
    float vals[cKMAX + 1]; int idxs[cKMAX + 1];
    #pragma unroll
    for (int k = 0; k <= cKMAX; ++k) { vals[k] = 1e30f; idxs[k] = 0; }
    for (int j = 0; j < cN; ++j) {
      const float v = dist[t * cN + j];
      if (v < vals[cKMAX]) {
        int p = cKMAX;
        while (p > 0 && v < vals[p - 1]) { vals[p] = vals[p - 1]; idxs[p] = idxs[p - 1]; --p; }
        vals[p] = v; idxs[p] = j;
      }
    }
    #pragma unroll
    for (int k = 0; k <= cKMAX; ++k) tkidx[t * (cKMAX + 1) + k] = idxs[k];
  }
}

// ---------------------------------------------------------------------------
// 2) nbr: per-token neighbor list = batch top list minus self, first K
// ---------------------------------------------------------------------------
__global__ __launch_bounds__(256) void nbr_kernel(
    const int* __restrict__ tkidx, const int* __restrict__ Kptr,
    int* __restrict__ nbr) {
  const int bn = blockIdx.x * 256 + threadIdx.x;
  if (bn >= cM) return;
  const int b = bn >> 9, n = bn & (cN - 1);
  const int K = *Kptr;
  int w = 0;
  #pragma unroll
  for (int k = 0; k <= cKMAX; ++k) {
    if (w >= K) break;
    const int j = tkidx[b * (cKMAX + 1) + k];
    if (j == n) continue;
    nbr[bn * cKMAX + (w++)] = j;
  }
}

// ---------------------------------------------------------------------------
// 3) GEMM: C[M,Nout] = A[M,Kd] @ W[Nout,Kd]^T, A/W pre-converted f16.
//    Workgroup (8 wave32) = 128x128 C macro-tile; wave = 16(M) x 128(N) strip.
//    B tile staged in LDS via global_load_async_to_lds_b128 (ASYNCcnt),
//    A fragment straight from global f16 (b128 x2), 8 v_wmma per k-step.
// ---------------------------------------------------------------------------
__global__ __launch_bounds__(256) void gemm_f16_kernel(
    const _Float16* __restrict__ A, const _Float16* __restrict__ W,
    float* __restrict__ C, int M, int Nout, int Kd) {
  __shared__ _Float16 Bs[128][40];          // rows padded 32->40 halves (banks)

  const int t    = threadIdx.x;
  const int lane = t & 31;
  const int wave = t >> 5;
  const int tilesN = Nout >> 7;             // N/128
  const int tm = blockIdx.x / tilesN, tn = blockIdx.x % tilesN;
  const int mi = lane & 15, hf = lane >> 4;

  const _Float16* __restrict__ arow = A + (size_t)(tm * 128 + wave * 16 + mi) * Kd;

  // cooperative B-tile mapping: thread -> (row, 32B segment)
  const int brow = t >> 1;
  const int bseg = t & 1;
  const _Float16* __restrict__ bsrc = W + (size_t)(tn * 128 + brow) * Kd + bseg * 16;
  const unsigned lds_dst = (unsigned)(size_t)(&Bs[brow][bseg * 16]);

  v8f acc[8] = {};

  for (int k0 = 0; k0 < Kd; k0 += 32) {
    // ---- async copy B tile (8KB) into LDS: 2 x b128 per thread ----
    const unsigned long long ga = (unsigned long long)(bsrc + k0);
    asm volatile("global_load_async_to_lds_b128 %0, %1, off"
                 :: "v"(lds_dst), "v"(ga) : "memory");
    asm volatile("global_load_async_to_lds_b128 %0, %1, off offset:16"
                 :: "v"(lds_dst), "v"(ga) : "memory");

    __builtin_prefetch(arow + k0 + 32, 0, 0);

    // ---- A fragment (contiguous k = 16*hf + s, same order as B from LDS) ----
    const v8h a0 = *(const v8h*)(arow + k0 + 16 * hf);       // k = 16*hf + e
    const v8h a1 = *(const v8h*)(arow + k0 + 16 * hf + 8);  // k = 16*hf + 8 + e
    v16h a;
    #pragma unroll
    for (int e = 0; e < 8; ++e) { a[e] = a0[e]; a[e + 8] = a1[e]; }

    asm volatile("s_wait_asynccnt 0" ::: "memory");
    __syncthreads();

    // ---- 8 WMMAs: one A frag x 8 B frags from LDS ----
    #pragma unroll
    for (int nt = 0; nt < 8; ++nt) {
      const v16h b = *(const v16h*)(&Bs[nt * 16 + mi][hf * 16]);  // k=16*hf+e, n=mi
      acc[nt] = __builtin_amdgcn_wmma_f32_16x16x32_f16(
          false, a, false, b, (short)0, acc[nt], false, false);
    }
    __syncthreads();   // protect Bs before next async fill
  }

  // C/D f32 layout: VGPR r, lanes 0-15 -> M=r, lanes 16-31 -> M=r+8
  #pragma unroll
  for (int nt = 0; nt < 8; ++nt) {
    float* __restrict__ crow = C + (size_t)(tm * 128 + wave * 16) * Nout
                                 + tn * 128 + nt * 16 + mi;
    #pragma unroll
    for (int r = 0; r < 8; ++r)
      crow[(size_t)(r + 8 * hf) * Nout] = acc[nt][r];
  }
}

// ---------------------------------------------------------------------------
// 4) ego Q fixup: rows flagged by ego_mask use Weq (fp32 exact, tiny work)
// ---------------------------------------------------------------------------
__global__ __launch_bounds__(256) void egoq_kernel(
    const float* __restrict__ tokens, const unsigned char* __restrict__ mask,
    const float* __restrict__ Weq, float* __restrict__ Qp) {
  const int b = blockIdx.x, t = threadIdx.x;
  for (int n = 0; n < cN; ++n) {
    if (!mask[b * cN + n]) continue;
    const float* __restrict__ x = tokens + ((size_t)b * cN + n) * cD;
    for (int o = t; o < cD; o += 256) {
      const float* __restrict__ w = Weq + (size_t)o * cD;
      float acc = 0.f;
      for (int d = 0; d < cD; ++d) acc = fmaf(x[d], w[d], acc);
      Qp[((size_t)b * cN + n) * cD + o] = acc;
    }
  }
}

// ---------------------------------------------------------------------------
// 5) bias MLP: relu([d_i,d_j] @ W1^T + b1) @ W2^T + b2  per (b,n,k,h)
// ---------------------------------------------------------------------------
__global__ __launch_bounds__(256) void bias_kernel(
    const float* __restrict__ dist, const int* __restrict__ nbr,
    const int* __restrict__ Kptr,
    const float* __restrict__ W1, const float* __restrict__ b1,
    const float* __restrict__ W2, const float* __restrict__ b2,
    float* __restrict__ bias) {
  const int gid = blockIdx.x * 256 + threadIdx.x;
  if (gid >= cM * cKMAX * cH) return;
  const int h  = gid & (cH - 1);
  const int k  = (gid >> 4) % cKMAX;
  const int bn = gid / (cKMAX * cH);
  const int K  = *Kptr;
  if (k >= K) return;
  const int b  = bn >> 9, n = bn & (cN - 1);
  const int j  = nbr[bn * cKMAX + k];
  const float di = dist[b * cN + n];
  const float dj = dist[b * cN + j];
  float acc = b2[h];
  for (int m = 0; m < cDM; ++m) {
    const float hid = fmaxf(fmaf(W1[m * 2 + 0], di, fmaf(W1[m * 2 + 1], dj, b1[m])), 0.f);
    acc = fmaf(W2[h * cDM + m], hid, acc);
  }
  bias[((size_t)bn * cKMAX + k) * cH + h] = acc;
}

// ---------------------------------------------------------------------------
// 6) attention core: block per (b,n); 256 thr = (h 0..15) x (i 0..15)
//    writes f16 directly (input of the final WMMA GEMM)
// ---------------------------------------------------------------------------
__global__ __launch_bounds__(256) void attn_kernel(
    const float* __restrict__ Qp, const float* __restrict__ Kp,
    const float* __restrict__ Vp, const float* __restrict__ bias,
    const int* __restrict__ nbr, const int* __restrict__ Kptr,
    _Float16* __restrict__ outp) {
  const int bn = blockIdx.x;
  const int b  = bn >> 9;
  const int t  = threadIdx.x;
  const int K  = *Kptr;

  __shared__ float Qs[cD];
  __shared__ float attn_s[cH][cKMAX];
  __shared__ float sc[cH][cKMAX];
  __shared__ int   js[cKMAX];

  for (int i = t; i < cD; i += 256) Qs[i] = Qp[(size_t)bn * cD + i];
  if (t < K) js[t] = nbr[bn * cKMAX + t];
  __syncthreads();

  const int h = t >> 4, i = t & 15;

  for (int k = 0; k < K; ++k) {
    const float* __restrict__ krow = Kp + ((size_t)b * cN + js[k]) * cD + h * cHD;
    float p = 0.f;
    #pragma unroll
    for (int d = 0; d < 4; ++d) p = fmaf(Qs[h * cHD + i * 4 + d], krow[i * 4 + d], p);
    #pragma unroll
    for (int off = 8; off > 0; off >>= 1) p += __shfl_xor(p, off, 16);
    if (i == 0)
      sc[h][k] = p * 0.125f + bias[((size_t)bn * cKMAX + k) * cH + h];
  }
  __syncthreads();

  if (i == 0) {
    float mx = -1e30f;
    for (int k = 0; k < K; ++k) mx = fmaxf(mx, sc[h][k]);
    float s = 0.f;
    for (int k = 0; k < K; ++k) { const float e = __expf(sc[h][k] - mx); attn_s[h][k] = e; s += e; }
    const float inv = 1.f / s;
    for (int k = 0; k < K; ++k) attn_s[h][k] *= inv;
  }
  __syncthreads();

  #pragma unroll
  for (int d = 0; d < 4; ++d) {
    float acc = 0.f;
    for (int k = 0; k < K; ++k)
      acc = fmaf(attn_s[h][k], Vp[((size_t)b * cN + js[k]) * cD + h * cHD + i * 4 + d], acc);
    outp[(size_t)bn * cD + h * cHD + i * 4 + d] = (_Float16)acc;
  }
}

// ---------------------------------------------------------------------------
// launch
// ---------------------------------------------------------------------------
extern "C" void kernel_launch(void* const* d_in, const int* in_sizes, int n_in,
                              void* d_out, int out_size, void* d_ws, size_t ws_size,
                              hipStream_t stream) {
  const float*         tokens = (const float*)d_in[0];
  const float*         dist   = (const float*)d_in[1];
  const unsigned char* mask   = (const unsigned char*)d_in[2];
  const float*         speed  = (const float*)d_in[3];
  const float*         Wq     = (const float*)d_in[4];
  const float*         Wk     = (const float*)d_in[5];
  const float*         Wv     = (const float*)d_in[6];
  const float*         Weq    = (const float*)d_in[7];
  const float*         Wo     = (const float*)d_in[8];
  const float*         W1     = (const float*)d_in[9];
  const float*         b1     = (const float*)d_in[10];
  const float*         W2     = (const float*)d_in[11];
  const float*         b2     = (const float*)d_in[12];

  char* ws = (char*)d_ws;
  int*      Kptr  = (int*)(ws + OFF_K);
  int*      tkidx = (int*)(ws + OFF_TK);
  int*      nbr   = (int*)(ws + OFF_NBR);
  float*    biasb = (float*)(ws + OFF_BIAS);
  float*    Qp    = (float*)(ws + OFF_QP);
  float*    Kp    = (float*)(ws + OFF_KP);
  float*    Vp    = (float*)(ws + OFF_VP);
  _Float16* Ah    = (_Float16*)(ws + OFF_AH);
  _Float16* Wqh   = (_Float16*)(ws + OFF_WQH);
  _Float16* Wkh   = (_Float16*)(ws + OFF_WKH);
  _Float16* Wvh   = (_Float16*)(ws + OFF_WVH);
  _Float16* Woh   = (_Float16*)(ws + OFF_WOH);
  _Float16* Aoh   = (_Float16*)(ws + OFF_AOH);
  float*    outp  = (float*)d_out;

  // 1) K + per-batch top-13 ; 2) neighbor lists
  meta_kernel<<<1, 256, 0, stream>>>(dist, speed, Kptr, tkidx);
  nbr_kernel<<<(cM + 255) / 256, 256, 0, stream>>>(tkidx, Kptr, nbr);

  // 0) one-shot f16 conversions
  const int nTok = cM * cD, nW = cD * cD;
  f32_to_f16_kernel<<<nTok / 2048, 256, 0, stream>>>(tokens, Ah, nTok);
  f32_to_f16_kernel<<<nW / 2048, 256, 0, stream>>>(Wq, Wqh, nW);
  f32_to_f16_kernel<<<nW / 2048, 256, 0, stream>>>(Wk, Wkh, nW);
  f32_to_f16_kernel<<<nW / 2048, 256, 0, stream>>>(Wv, Wvh, nW);
  f32_to_f16_kernel<<<nW / 2048, 256, 0, stream>>>(Wo, Woh, nW);

  // 3) projections: 128x128 macro-tiles, grid = (M/128)*(D/128) = 256
  const int ggrid = (cM / 128) * (cD / 128);
  gemm_f16_kernel<<<ggrid, 256, 0, stream>>>(Ah, Wqh, Qp, cM, cD, cD);
  gemm_f16_kernel<<<ggrid, 256, 0, stream>>>(Ah, Wkh, Kp, cM, cD, cD);
  gemm_f16_kernel<<<ggrid, 256, 0, stream>>>(Ah, Wvh, Vp, cM, cD, cD);

  // 4) ego rows use Weq (fp32 exact)
  egoq_kernel<<<cB, 256, 0, stream>>>(tokens, mask, Weq, Qp);

  // 5) additive distance-bias MLP
  bias_kernel<<<(cM * cKMAX * cH) / 256, 256, 0, stream>>>(
      dist, nbr, Kptr, W1, b1, W2, b2, biasb);

  // 6) attention core -> f16
  attn_kernel<<<cM, 256, 0, stream>>>(Qp, Kp, Vp, biasb, nbr, Kptr, Aoh);

  // 7) output projection
  gemm_f16_kernel<<<ggrid, 256, 0, stream>>>(Aoh, Woh, outp, cM, cD, cD);
}
